// part_iwt1_20315195310501
// MI455X (gfx1250) — matrix operand order repaired and checked by reading, not compiled, not run on hardware
//
#include <hip/hip_runtime.h>
#include <hip/hip_fp16.h>

typedef __attribute__((ext_vector_type(16))) _Float16 v16h;
typedef __attribute__((ext_vector_type(8)))  _Float16 v8h;
typedef __attribute__((ext_vector_type(8)))  float    v8f;
typedef __attribute__((ext_vector_type(4)))  int      v4i;

#define AS_GLOBAL __attribute__((address_space(1)))
#define AS_LDS    __attribute__((address_space(3)))

#if __has_builtin(__builtin_amdgcn_global_load_async_to_lds_b128)
#define HAVE_ASYNC_LDS 1
#endif

// 16B global -> LDS copy (async on CDNA5; sync fallback otherwise)
__device__ inline void cp16(_Float16* ldst, const _Float16* gsrc) {
#ifdef HAVE_ASYNC_LDS
  __builtin_amdgcn_global_load_async_to_lds_b128(
      (AS_GLOBAL v4i*)const_cast<_Float16*>(gsrc),
      (AS_LDS v4i*)ldst, 0, 0);
#else
  *(v8h*)ldst = *(const v8h*)gsrc;
#endif
}

__device__ inline void cp_wait() {
#ifdef HAVE_ASYNC_LDS
#if __has_builtin(__builtin_amdgcn_s_wait_asynccnt)
  __builtin_amdgcn_s_wait_asynccnt(0);
#endif
#endif
  __syncthreads();
}

// ---------------------------------------------------------------------------
// WMMA fragment helpers (gfx1250, wave32, 16x16x32 f16 -> f32)
// A (16x32 MxK): lane L (row=L&15, half=L>>4): K = e + 16*half
//                (two 16B K-contiguous loads)
// B (32x16 KxN): lane L (col=L&15, half=L>>4): K = e + 16*half
// ---------------------------------------------------------------------------
__device__ inline v16h frag_a(const _Float16* base, int rowStrideHalves) {
  int lane = threadIdx.x & 31;
  int r = lane & 15, h = lane >> 4;
  const _Float16* p = base + (size_t)r * rowStrideHalves + h * 16;
  v8h lo = *(const v8h*)(p);
  v8h hi = *(const v8h*)(p + 8);
  v16h out;
#pragma unroll
  for (int i = 0; i < 8; ++i) { out[i] = lo[i]; out[i + 8] = hi[i]; }
  return out;
}

__device__ inline v16h frag_b(const _Float16* base, int colStrideHalves) {
  int lane = threadIdx.x & 31;
  int c = lane & 15, h = lane >> 4;
  const _Float16* p = base + (size_t)c * colStrideHalves + h * 16;
  v8h lo = *(const v8h*)(p);
  v8h hi = *(const v8h*)(p + 8);
  v16h out;
#pragma unroll
  for (int i = 0; i < 8; ++i) { out[i] = lo[i]; out[i + 8] = hi[i]; }
  return out;
}

__device__ inline v8f wmma16(v16h a, v16h b, v8f c) {
  return __builtin_amdgcn_wmma_f32_16x16x32_f16(false, a, false, b, (short)0, c,
                                                false, false);
}

// ---------------------------------------------------------------------------
// Weight / activation packing
// ---------------------------------------------------------------------------
__global__ void k_pack_w3(const float* __restrict__ w, _Float16* __restrict__ out) {
  int i = blockIdx.x * 256 + threadIdx.x;           // 128*128*9
  if (i >= 128 * 128 * 9) return;
  int tap = i % 9; int ci = (i / 9) & 127; int co = i / (9 * 128);
  out[(size_t)co * 1152 + tap * 128 + ci] = (_Float16)w[i];
}

__global__ void k_pack_up(const float* __restrict__ w, _Float16* __restrict__ out) {
  int i = blockIdx.x * 256 + threadIdx.x;           // 128*256*4
  if (i >= 128 * 256 * 4) return;
  int tap = i & 3; int ci = (i >> 2) & 255; int co = i >> 10;
  out[((size_t)tap * 128 + co) * 256 + ci] = (_Float16)w[i];
}

__global__ void k_cast_f16(const float* __restrict__ in, _Float16* __restrict__ out, int n) {
  int i = blockIdx.x * 256 + threadIdx.x;
  if (i < n) out[i] = (_Float16)in[i];
}

__global__ void k_x1h(const float* __restrict__ x1, _Float16* __restrict__ out) {
  int i = blockIdx.x * 256 + threadIdx.x;           // 16*256*64*64 NCHW index
  if (i >= 16 * 256 * 64 * 64) return;
  int x = i & 63; int y = (i >> 6) & 63; int c = (i >> 12) & 255; int n = i >> 20;
  out[(((size_t)n * 64 + y) * 64 + x) * 256 + c] = (_Float16)x1[i];
}

// fold BN (+optional conv bias) into scale/bias:  y = x*s + b
__global__ void k_sb(const float* __restrict__ bn4, const float* __restrict__ cb,
                     int C, float* __restrict__ out) {
  int c = blockIdx.x * 64 + threadIdx.x;
  if (c >= C) return;
  float s = bn4[c] * rsqrtf(bn4[3 * C + c] + 1e-5f);
  float cbv = cb ? cb[c] : 0.f;
  out[c] = s;
  out[C + c] = (cbv - bn4[2 * C + c]) * s + bn4[C + c];
}

// ---------------------------------------------------------------------------
// Transposed conv 2x2 (lhs_dilation=2, pad=1): exactly one tap per output.
// Implicit GEMM: M = 16 outputs of one parity, K = 256, N = 16/wave.
// A tile (16 px x 256 ch = 8KB) staged to LDS via async copy, shared by waves.
// ---------------------------------------------------------------------------
__global__ __launch_bounds__(256) void k_up(const _Float16* __restrict__ x1h,
                                            const _Float16* __restrict__ wup,
                                            const float* __restrict__ upb,
                                            float* __restrict__ xh) {
  __shared__ _Float16 tile[16 * 256];               // 8 KB
  int wave = threadIdx.x >> 5;                      // ntile 0..7
  int mt = blockIdx.x;                              // 16384 = n*128p*2px*4qxb
  int qxb = mt & 3; int px = (mt >> 2) & 1; int p = (mt >> 3) & 127; int n = mt >> 10;
  int py = p & 1, qy = p >> 1;
  int tap = (1 - py) * 2 + (1 - px);
  int qx0 = qxb * 16;
  const _Float16* abase = x1h + ((size_t)(n * 64 + qy) * 64 + qx0) * 256;
  // cooperative stage: contiguous 4096 halves
#pragma unroll
  for (int i = 0; i < 2; ++i) {
    int j = (threadIdx.x + i * 256) * 8;
    cp16(&tile[j], abase + j);
  }
  cp_wait();
  const _Float16* wb = wup + ((size_t)tap * 128 + wave * 16) * 256;
  __builtin_prefetch(wb, 0, 1);
  v8f acc = {};
#pragma unroll
  for (int kc = 0; kc < 8; ++kc)
    acc = wmma16(frag_a(&tile[kc * 32], 256), frag_b(wb + kc * 32, 256), acc);
  int lane = threadIdx.x & 31;
  int co = wave * 16 + (lane & 15);
  float bb = upb[co];
#pragma unroll
  for (int v = 0; v < 8; ++v) {
    int m = v + 8 * (lane >> 4);
    int q = 2 * (qx0 + m) + px;
    xh[((size_t)(n * 128 + p) * 128 + q) * 128 + co] = acc[v] + bb;
  }
}

// ---------------------------------------------------------------------------
// Inverse wavelet grouped conv (4 subbands, effective 3x3 taps). Scalar.
// ---------------------------------------------------------------------------
__global__ __launch_bounds__(256) void k_iwt(const float* __restrict__ x2,
                                             float* __restrict__ xlf,
                                             _Float16* __restrict__ xlh) {
  const float lo[6] = {-0.01565572813546454f, -0.0727326195128539f, 0.38486484686420286f,
                       0.8525720202122554f,   0.3378976624578092f,  -0.0727326195128539f};
  const float hi[6] = {0.0727326195128539f,  0.3378976624578092f,  -0.8525720202122554f,
                       0.38486484686420286f, 0.0727326195128539f,  -0.01565572813546454f};
  size_t idx = (size_t)blockIdx.x * 256 + threadIdx.x;   // (n,c,p,q) order
  if (idx >= (size_t)16 * 128 * 128 * 128) return;
  int q = idx & 127; int p = (idx >> 7) & 127; int c = (idx >> 14) & 127; int n = idx >> 21;
  int i0 = 1 - (p & 1), j0 = 1 - (q & 1);
  int ry = p >> 1, rx = q >> 1;
  const float* xb = x2 + ((size_t)(n * 512 + 4 * c)) * 66 * 66;
  float acc = 0.f;
#pragma unroll
  for (int s = 0; s < 4; ++s) {
    const float* wi = (s & 1) ? hi : lo;
    const float* wj = (s & 2) ? hi : lo;
    const float* sb = xb + (size_t)s * 66 * 66;
#pragma unroll
    for (int t = 0; t < 3; ++t)
#pragma unroll
      for (int u = 0; u < 3; ++u)
        acc += wi[i0 + 2 * t] * wj[j0 + 2 * u] * sb[(ry + t) * 66 + (rx + u)];
  }
  size_t o = (((size_t)(n * 128 + p) * 128 + q) * 128 + c);
  xlf[o] = acc;
  xlh[o] = (_Float16)acc;
}

// ---------------------------------------------------------------------------
// Per-image channel mean of xh (NHWC)
// ---------------------------------------------------------------------------
__global__ __launch_bounds__(256) void k_gmean(const float* __restrict__ xh,
                                               float* __restrict__ g) {
  __shared__ float red[256];
  int n = blockIdx.x; int c = threadIdx.x & 127; int pg = threadIdx.x >> 7;
  const float* base = xh + (size_t)n * 16384 * 128;
  float s = 0.f;
  for (int pix = pg; pix < 16384; pix += 2) s += base[(size_t)pix * 128 + c];
  red[threadIdx.x] = s;
  __syncthreads();
  if (pg == 0) g[n * 128 + c] = (red[c] + red[128 + c]) * (1.f / 16384.f);
}

// ---------------------------------------------------------------------------
// Channel-attention MLP (tiny; scalar f32)
// ---------------------------------------------------------------------------
__global__ __launch_bounds__(128) void k_td(const float* __restrict__ g,
                                            const float* __restrict__ w1,
                                            const float* __restrict__ w2,
                                            const float* __restrict__ sb1,
                                            const float* __restrict__ sb2,
                                            float* __restrict__ td) {
  __shared__ float gs[128], mid[32];
  int n = blockIdx.x, t = threadIdx.x;
  gs[t] = g[n * 128 + t];
  __syncthreads();
  if (t < 32) {
    float a = 0.f;
    for (int i = 0; i < 128; ++i) a += w1[t * 128 + i] * gs[i];
    a = a * sb1[t] + sb1[32 + t];
    mid[t] = a > 0.f ? a : 0.f;
  }
  __syncthreads();
  float a = 0.f;
  for (int i = 0; i < 32; ++i) a += w2[t * 32 + i] * mid[i];
  a = a * sb2[t] + sb2[128 + t];
  td[n * 128 + t] = 1.f / (1.f + __expf(-a));
}

// ---------------------------------------------------------------------------
// Fused bu-path + combine:  mid = relu(bn1(xl@W1)) ; bu = sigmoid(bn2(mid@W2))
// xs = 2*(xl*td + xh*bu)   -> padded NHWC f16
// ---------------------------------------------------------------------------
__global__ __launch_bounds__(256) void k_xs(const _Float16* __restrict__ xlh,
                                            const float* __restrict__ xlf,
                                            const float* __restrict__ xhf,
                                            const _Float16* __restrict__ w1,
                                            const _Float16* __restrict__ w2,
                                            const float* __restrict__ sb1,
                                            const float* __restrict__ sb2,
                                            const float* __restrict__ td,
                                            _Float16* __restrict__ xs) {
  __shared__ _Float16 mid[8][16 * 32];
  int wave = threadIdx.x >> 5, lane = threadIdx.x & 31;
  size_t pix0 = (size_t)blockIdx.x * 128 + wave * 16;
  const _Float16* abase = xlh + pix0 * 128;

  v8f acc0 = {}, acc1 = {};
#pragma unroll
  for (int kc = 0; kc < 4; ++kc) {
    v16h a = frag_a(abase + kc * 32, 128);
    acc0 = wmma16(a, frag_b(w1 + kc * 32, 128), acc0);
    acc1 = wmma16(a, frag_b(w1 + 16 * 128 + kc * 32, 128), acc1);
  }
  {
    int c = lane & 15, hh = lane >> 4;
    float s0 = sb1[c], b0 = sb1[32 + c], s1 = sb1[16 + c], b1 = sb1[48 + c];
#pragma unroll
    for (int v = 0; v < 8; ++v) {
      int m = v + 8 * hh;
      float e0 = acc0[v] * s0 + b0; e0 = e0 > 0.f ? e0 : 0.f;
      float e1 = acc1[v] * s1 + b1; e1 = e1 > 0.f ? e1 : 0.f;
      mid[wave][m * 32 + c]      = (_Float16)e0;
      mid[wave][m * 32 + 16 + c] = (_Float16)e1;
    }
  }
  v16h a2 = frag_a(&mid[wave][0], 32);   // same-wave produce/consume
#pragma unroll
  for (int nt = 0; nt < 8; ++nt) {
    v8f acc = {};
    acc = wmma16(a2, frag_b(w2 + nt * 16 * 32, 32), acc);
    int co = nt * 16 + (lane & 15);
    float s2 = sb2[co], b2 = sb2[128 + co], tv;
#pragma unroll
    for (int v = 0; v < 8; ++v) {
      int m = v + 8 * (lane >> 4);
      size_t pix = pix0 + m;
      float bu = 1.f / (1.f + __expf(-(acc[v] * s2 + b2)));
      int n = (int)(pix >> 14); int p = (int)((pix >> 7) & 127); int q = (int)(pix & 127);
      tv = td[n * 128 + co];
      float val = 2.f * (xlf[pix * 128 + co] * tv + xhf[pix * 128 + co] * bu);
      xs[((size_t)(n * 130 + p + 1) * 130 + (q + 1)) * 128 + co] = (_Float16)val;
    }
  }
}

// ---------------------------------------------------------------------------
// 3x3 conv 128->128, pad 1, fused BN+ReLU. Implicit GEMM, K = 9*128.
// Input tile (3 rows x 18 px x 128 ch, 13.8 KB) staged to LDS with async
// global->LDS copies, shared by all 8 waves; B fragments from L2-resident
// packed weights. Output: padded f16 interior (outH) or f32 NCHW (outF).
// ---------------------------------------------------------------------------
__global__ __launch_bounds__(256) void k_conv3(const _Float16* __restrict__ in,
                                               const _Float16* __restrict__ w,
                                               const float* __restrict__ sb,
                                               _Float16* __restrict__ outH,
                                               float* __restrict__ outF) {
  __shared__ _Float16 tile[3 * 18 * 128];           // 6912 halves = 13824 B
  int wave = threadIdx.x >> 5;                      // ntile 0..7
  int mt = blockIdx.x;                              // 16384 = 16n * 128y * 8tx
  int tx = mt & 7; int y = (mt >> 3) & 127; int n = mt >> 10;
  int x0 = tx * 16;
  // cooperative async stage: rows y..y+2, px x0..x0+17, all 128 channels
  {
    const int ROWH = 18 * 128;                      // halves per row
    for (int i = threadIdx.x; i < (3 * ROWH) / 8; i += 256) {
      int j = i * 8;
      int row = j / ROWH;
      int rem = j - row * ROWH;
      const _Float16* g = in + ((size_t)(n * 130 + y + row) * 130 + x0) * 128 + rem;
      cp16(&tile[row * ROWH + rem], g);
    }
  }
  cp_wait();
  const _Float16* wb = w + (size_t)(wave * 16) * 1152;
  __builtin_prefetch(wb, 0, 1);
  v8f acc = {};
#pragma unroll
  for (int dy = 0; dy < 3; ++dy)
#pragma unroll
    for (int dx = 0; dx < 3; ++dx) {
      const _Float16* ab = &tile[(dy * 18 + dx) * 128];
      int kb = (dy * 3 + dx) * 128;
#pragma unroll
      for (int kc = 0; kc < 4; ++kc)
        acc = wmma16(frag_a(ab + kc * 32, 128), frag_b(wb + kb + kc * 32, 1152), acc);
    }
  int lane = threadIdx.x & 31;
  int co = wave * 16 + (lane & 15);
  float s = sb[co], bb = sb[128 + co];
#pragma unroll
  for (int v = 0; v < 8; ++v) {
    int m = v + 8 * (lane >> 4);
    float val = acc[v] * s + bb;
    val = val > 0.f ? val : 0.f;
    int x = x0 + m;
    if (outH)
      outH[((size_t)(n * 130 + y + 1) * 130 + (x + 1)) * 128 + co] = (_Float16)val;
    else
      outF[(((size_t)n * 128 + co) * 128 + y) * 128 + x] = val;
  }
}

// ---------------------------------------------------------------------------
extern "C" void kernel_launch(void* const* d_in, const int* in_sizes, int n_in,
                              void* d_out, int out_size, void* d_ws, size_t ws_size,
                              hipStream_t stream) {
  (void)in_sizes; (void)n_in; (void)out_size; (void)ws_size;
  const float* x1     = (const float*)d_in[0];
  const float* x2     = (const float*)d_in[1];
  const float* up_w   = (const float*)d_in[2];
  const float* up_b   = (const float*)d_in[3];
  const float* td_w1  = (const float*)d_in[4];
  const float* td_b1  = (const float*)d_in[5];
  const float* td_bn1 = (const float*)d_in[6];
  const float* td_w2  = (const float*)d_in[7];
  const float* td_b2  = (const float*)d_in[8];
  const float* td_bn2 = (const float*)d_in[9];
  const float* bu_w1  = (const float*)d_in[10];
  const float* bu_b1  = (const float*)d_in[11];
  const float* bu_bn1 = (const float*)d_in[12];
  const float* bu_w2  = (const float*)d_in[13];
  const float* bu_b2  = (const float*)d_in[14];
  const float* bu_bn2 = (const float*)d_in[15];
  const float* post_w = (const float*)d_in[16];
  const float* post_b = (const float*)d_in[17];
  const float* post_bn= (const float*)d_in[18];
  const float* dc_w1  = (const float*)d_in[19];
  const float* dc_bn1 = (const float*)d_in[20];
  const float* dc_w2  = (const float*)d_in[21];
  const float* dc_bn2 = (const float*)d_in[22];
  float* y_out = (float*)d_out;

  char* ws = (char*)d_ws;
  size_t off = 0;
  auto take = [&](size_t sz) { char* p = ws + off; off = (off + sz + 255) & ~(size_t)255; return p; };

  const size_t NPIX = (size_t)16 * 128 * 128;       // 262144 pixels
  float*    XH   = (float*)   take(NPIX * 128 * 4);           // xh f32 NHWC
  float*    XLF  = (float*)   take(NPIX * 128 * 4);           // xl f32 NHWC
  _Float16* XLH  = (_Float16*)take(NPIX * 128 * 2);           // xl f16 NHWC
  size_t padB = (size_t)16 * 130 * 130 * 128 * 2;
  _Float16* XSH  = (_Float16*)take(padB);                     // padded f16 buf A
  _Float16* T1H  = (_Float16*)take(padB);                     // padded f16 buf B
  _Float16* X1H  = (_Float16*)take((size_t)16 * 64 * 64 * 256 * 2);
  _Float16* WUP  = (_Float16*)take((size_t)4 * 128 * 256 * 2);
  _Float16* WPOST= (_Float16*)take((size_t)128 * 1152 * 2);
  _Float16* WDC1 = (_Float16*)take((size_t)128 * 1152 * 2);
  _Float16* WDC2 = (_Float16*)take((size_t)128 * 1152 * 2);
  _Float16* WBU1 = (_Float16*)take((size_t)32 * 128 * 2);
  _Float16* WBU2 = (_Float16*)take((size_t)128 * 32 * 2);
  float* SBTD1 = (float*)take(2 * 32 * 4);
  float* SBTD2 = (float*)take(2 * 128 * 4);
  float* SBBU1 = (float*)take(2 * 32 * 4);
  float* SBBU2 = (float*)take(2 * 128 * 4);
  float* SBPO  = (float*)take(2 * 128 * 4);
  float* SBD1  = (float*)take(2 * 128 * 4);
  float* SBD2  = (float*)take(2 * 128 * 4);
  float* G  = (float*)take(16 * 128 * 4);
  float* TD = (float*)take(16 * 128 * 4);

  // --- prep ---
  (void)hipMemsetAsync(XSH, 0, padB, stream);
  (void)hipMemsetAsync(T1H, 0, padB, stream);
  k_pack_w3<<<(128 * 128 * 9 + 255) / 256, 256, 0, stream>>>(post_w, WPOST);
  k_pack_w3<<<(128 * 128 * 9 + 255) / 256, 256, 0, stream>>>(dc_w1, WDC1);
  k_pack_w3<<<(128 * 128 * 9 + 255) / 256, 256, 0, stream>>>(dc_w2, WDC2);
  k_pack_up<<<(128 * 256 * 4 + 255) / 256, 256, 0, stream>>>(up_w, WUP);
  k_cast_f16<<<(4096 + 255) / 256, 256, 0, stream>>>(bu_w1, WBU1, 32 * 128);
  k_cast_f16<<<(4096 + 255) / 256, 256, 0, stream>>>(bu_w2, WBU2, 128 * 32);
  k_x1h<<<(16 * 256 * 64 * 64 + 255) / 256, 256, 0, stream>>>(x1, X1H);
  k_sb<<<1, 64, 0, stream>>>(td_bn1, td_b1, 32, SBTD1);
  k_sb<<<2, 64, 0, stream>>>(td_bn2, td_b2, 128, SBTD2);
  k_sb<<<1, 64, 0, stream>>>(bu_bn1, bu_b1, 32, SBBU1);
  k_sb<<<2, 64, 0, stream>>>(bu_bn2, bu_b2, 128, SBBU2);
  k_sb<<<2, 64, 0, stream>>>(post_bn, post_b, 128, SBPO);
  k_sb<<<2, 64, 0, stream>>>(dc_bn1, (const float*)nullptr, 128, SBD1);
  k_sb<<<2, 64, 0, stream>>>(dc_bn2, (const float*)nullptr, 128, SBD2);

  // --- front: xh (WMMA implicit GEMM) and xl (scalar IWT) ---
  k_up<<<16384, 256, 0, stream>>>(X1H, WUP, up_b, XH);
  k_iwt<<<(int)((NPIX * 128 + 255) / 256), 256, 0, stream>>>(x2, XLF, XLH);

  // --- attention ---
  k_gmean<<<16, 256, 0, stream>>>(XH, G);
  k_td<<<16, 128, 0, stream>>>(G, td_w1, td_w2, SBTD1, SBTD2, TD);

  // --- fused bu + combine -> xs (padded) ---
  k_xs<<<2048, 256, 0, stream>>>(XLH, XLF, XH, WBU1, WBU2, SBBU1, SBBU2, TD, XSH);

  // --- 3x3 conv chain ---
  k_conv3<<<16384, 256, 0, stream>>>(XSH, WPOST, SBPO, T1H, (float*)nullptr);
  k_conv3<<<16384, 256, 0, stream>>>(T1H, WDC1, SBD1, XSH, (float*)nullptr);
  k_conv3<<<16384, 256, 0, stream>>>(XSH, WDC2, SBD2, (_Float16*)nullptr, y_out);
}
